// Encoder_5360119185930
// MI455X (gfx1250) — compile-verified, matrix operand order checked by reading, not run on hardware
//
#include <hip/hip_runtime.h>
#include <cstdint>
#include <cstddef>

typedef __attribute__((ext_vector_type(16))) __bf16 v16bf;
typedef __attribute__((ext_vector_type(8)))  float  v8f;
typedef unsigned short u16;

namespace {

constexpr int kB      = 16;
constexpr int kLin    = 512;
constexpr int kD      = 512;
constexpr int kHeads  = 8;
constexpr int kDh     = 128;
constexpr int kDff    = 2048;
constexpr int kLreal  = 680;          // 512+128+32+8
constexpr int kLpad   = 704;          // 22*32, 44*16
constexpr int kRows   = kB * kLpad;   // 11264
constexpr int kPyrL   = 168;          // 128+32+8

__device__ __forceinline__ u16 f2bf(float f) {
  unsigned u = __builtin_bit_cast(unsigned, f);
  return (u16)((u + 0x7FFFu + ((u >> 16) & 1u)) >> 16);
}
__device__ __forceinline__ float bf2f(u16 h) {
  unsigned u = ((unsigned)h) << 16;
  return __builtin_bit_cast(float, u);
}

// ---------------------------------------------------------------- mask bias
__device__ __forceinline__ int lev_of(int p) {
  return p < 512 ? 0 : p < 640 ? 1 : p < 672 ? 2 : 3;
}
__device__ __forceinline__ bool child_vis(int par, int lp, int ch) {
  const int st[4] = {0, 512, 640, 672};
  const int sz[4] = {512, 128, 32, 8};
  int s  = st[lp];
  int lo = st[lp - 1] + (par - s) * 4;
  int hi = (par == s + sz[lp] - 1) ? s : st[lp - 1] + (par - s + 1) * 4;
  return ch >= lo && ch < hi;
}
__global__ __launch_bounds__(256) void build_bias(float* __restrict__ bias) {
  long i = (long)blockIdx.x * 256 + threadIdx.x;
  if (i >= (long)kLpad * kLpad) return;
  int jj = (int)(i % kLpad), ii = (int)(i / kLpad);
  float v = -1e9f;
  if (ii < kLreal && jj < kLreal) {
    int li = lev_of(ii), lj = lev_of(jj);
    bool vis = false;
    if (li == lj) { int d = ii - jj; vis = (d <= 2 && d >= -2); }
    else if (li == lj + 1) vis = child_vis(ii, li, jj);
    else if (lj == li + 1) vis = child_vis(jj, lj, ii);
    if (vis) v = 0.f;
  }
  bias[i] = v;
}

// ------------------------------------------------- weight transpose+convert
// dst[n*K + k] = bf16(src[k*N + n])
__global__ __launch_bounds__(256) void wtrans(const float* __restrict__ src,
                                              u16* __restrict__ dst, int K, int N) {
  long i = (long)blockIdx.x * 256 + threadIdx.x;
  if (i >= (long)K * N) return;
  long n = i / K, kk = i % K;
  dst[i] = f2bf(src[kk * (long)N + n]);
}

// ---------------------------------------------------------------- embedding
__global__ __launch_bounds__(256) void embed_kernel(
    const float* __restrict__ x, const float* __restrict__ cov_w,
    const float* __restrict__ cov_b, const float* __restrict__ dcw,
    float* __restrict__ enc_f, u16* __restrict__ seq_h) {
  long i = (long)blockIdx.x * 256 + threadIdx.x;
  if (i >= (long)kB * kLin * kD) return;
  int d = (int)(i & 511); long bl = i >> 9;
  int l = (int)(bl & 511); int b = (int)(bl >> 9);
  const float* xr = x + ((long)b * kLin + l) * 6;
  float e = cov_b[d];
  #pragma unroll
  for (int c = 0; c < 4; ++c) e += xr[1 + c] * cov_w[c * kD + d];
  e += (xr[5] * (1.f / 128.f) - 0.5f) * cov_w[4 * kD + d];
  int lm = (l + 511) & 511, lp = (l + 1) & 511;
  float d0 = x[((long)b * kLin + lm) * 6];
  float d1 = xr[0];
  float d2 = x[((long)b * kLin + lp) * 6];
  e += d0 * dcw[d * 3 + 0] + d1 * dcw[d * 3 + 1] + d2 * dcw[d * 3 + 2];
  int dd = d >> 1;
  float vexp = __powf(10000.f, (2.f * dd) * (1.f / 512.f));
  float ang = (float)l / vexp;
  e += (d & 1) ? __cosf(ang) : __sinf(ang);
  enc_f[((long)b * kLpad + l) * kD + d] = e;
  seq_h[((long)b * kLin + l) * kD + d] = f2bf(e);
}

// ------------------------------------------------------------- WMMA GEMM
// C[M,N] = A(bf16,[M,K]) * Wt(bf16,[N,K])^T, one wave per 16x128 tile.
// EPI: 1 f32+bias | 2 bf16 | 3 bf16+bias | 4 f32+resid | 5 bf16+bias+gelu
//      6 f32+bias+resid | 7 bf16 scattered into Vt[b][h][d][pos]
template <int EPI>
__global__ __launch_bounds__(256) void gemm_bf16(
    const u16* __restrict__ A, const u16* __restrict__ Wt, void* __restrict__ outv,
    const float* __restrict__ bias, const float* __restrict__ resid,
    int M, int N, int K) {
  int lane = threadIdx.x & 31;
  int wave = threadIdx.x >> 5;
  int hl = lane >> 4, r = lane & 15;
  int tiles_n = N >> 7;
  long tile = (long)blockIdx.x * 8 + wave;
  long tiles = ((long)(M >> 4)) * tiles_n;
  if (tile >= tiles) return;
  int n0 = (int)(tile % tiles_n) << 7;
  long m0 = (tile / tiles_n) << 4;

  v8f acc[8] = {};
  const u16* arow = A + (m0 + r) * (long)K + hl * 16;
  for (int k0 = 0; k0 < K; k0 += 32) {
    v16bf a = *(const v16bf*)(const void*)(arow + k0);
    #pragma unroll
    for (int n = 0; n < 8; ++n) {
      const u16* bp = Wt + ((long)(n0 + n * 16 + r)) * K + k0 + hl * 16;
      v16bf bfr = *(const v16bf*)(const void*)bp;
      acc[n] = __builtin_amdgcn_wmma_f32_16x16x32_bf16(
          false, a, false, bfr, (short)0, acc[n], false, false);
    }
  }
  #pragma unroll
  for (int n = 0; n < 8; ++n) {
    int col = n0 + n * 16 + r;
    #pragma unroll
    for (int j = 0; j < 8; ++j) {
      long row = m0 + j + 8 * hl;
      float v = acc[n][j];
      if (EPI == 1 || EPI == 3 || EPI == 5 || EPI == 6) v += bias[col];
      if (EPI == 5) v = 0.5f * v * (1.0f + erff(v * 0.70710678118f));
      if (EPI == 4 || EPI == 6) v += resid[row * N + col];
      if (EPI == 2 || EPI == 3 || EPI == 5) {
        ((u16*)outv)[row * N + col] = f2bf(v);
      } else if (EPI == 7) {
        int b = (int)(row / kLpad), pos = (int)(row % kLpad);
        int hh = col >> 7, d = col & 127;
        ((u16*)outv)[(((long)(b * kHeads + hh)) * kDh + d) * kLpad + pos] = f2bf(v);
      } else {
        ((float*)outv)[row * N + col] = v;
      }
    }
  }
}

// -------------------------------------------------------- flash attention
__global__ __launch_bounds__(32) void fa_kernel(
    const u16* __restrict__ q, const u16* __restrict__ kmat,
    const u16* __restrict__ vt, const float* __restrict__ bias,
    u16* __restrict__ o) {
  alignas(32) __shared__ u16 plds[16 * 32];
  int lane = threadIdx.x & 31;
  int hl = lane >> 4, r = lane & 15;
  int id = blockIdx.x;
  int qt = id % (kLpad / 16); id /= (kLpad / 16);
  int h = id % kHeads; int b = id / kHeads;
  int q0 = qt * 16;
  const long rowstr = kHeads * kDh;  // 1024
  const u16* qbase = q + (long)b * kLpad * rowstr + h * kDh;
  const u16* kbase = kmat + (long)b * kLpad * rowstr + h * kDh;
  const u16* vbase = vt + ((long)(b * kHeads + h)) * kDh * kLpad;

  v16bf qa[4];
  #pragma unroll
  for (int c = 0; c < 4; ++c)
    qa[c] = *(const v16bf*)(const void*)(qbase + (long)(q0 + r) * rowstr + c * 32 + hl * 16);

  v8f oacc[8] = {};
  float rm[8], rs[8];
  #pragma unroll
  for (int j = 0; j < 8; ++j) { rm[j] = -3.0e38f; rs[j] = 0.f; }

  const float sc = 0.08838834764831845f;  // 1/sqrt(128)
  for (int kb = 0; kb < kLpad / 32; ++kb) {
    int kb0 = kb * 32;
    v8f s0 = {}, s1 = {};
    #pragma unroll
    for (int c = 0; c < 4; ++c) {
      v16bf kf0 = *(const v16bf*)(const void*)(kbase + (long)(kb0 + r) * rowstr + c * 32 + hl * 16);
      v16bf kf1 = *(const v16bf*)(const void*)(kbase + (long)(kb0 + 16 + r) * rowstr + c * 32 + hl * 16);
      s0 = __builtin_amdgcn_wmma_f32_16x16x32_bf16(false, qa[c], false, kf0, (short)0, s0, false, false);
      s1 = __builtin_amdgcn_wmma_f32_16x16x32_bf16(false, qa[c], false, kf1, (short)0, s1, false, false);
    }
    #pragma unroll
    for (int j = 0; j < 8; ++j) {
      int qg = q0 + j + 8 * hl;
      s0[j] = s0[j] * sc + bias[(long)qg * kLpad + kb0 + r];
      s1[j] = s1[j] * sc + bias[(long)qg * kLpad + kb0 + 16 + r];
    }
    #pragma unroll
    for (int j = 0; j < 8; ++j) {
      float t = fmaxf(s0[j], s1[j]);
      for (int off = 1; off < 16; off <<= 1) t = fmaxf(t, __shfl_xor(t, off, 16));
      float nm = fmaxf(rm[j], t);
      float corr = __expf(rm[j] - nm);
      rm[j] = nm;
      float p0 = __expf(s0[j] - nm);
      float p1 = __expf(s1[j] - nm);
      float ps = p0 + p1;
      for (int off = 1; off < 16; off <<= 1) ps += __shfl_xor(ps, off, 16);
      rs[j] = rs[j] * corr + ps;
      s0[j] = p0; s1[j] = p1;
      #pragma unroll
      for (int n = 0; n < 8; ++n) oacc[n][j] *= corr;
    }
    #pragma unroll
    for (int j = 0; j < 8; ++j) {
      plds[(j + 8 * hl) * 32 + r]      = f2bf(s0[j]);
      plds[(j + 8 * hl) * 32 + 16 + r] = f2bf(s1[j]);
    }
    __syncthreads();
    v16bf pa = *(const v16bf*)(const void*)(plds + r * 32 + hl * 16);
    __syncthreads();
    #pragma unroll
    for (int n = 0; n < 8; ++n) {
      v16bf vf = *(const v16bf*)(const void*)(vbase + (long)(n * 16 + r) * kLpad + kb0 + hl * 16);
      oacc[n] = __builtin_amdgcn_wmma_f32_16x16x32_bf16(false, pa, false, vf, (short)0, oacc[n], false, false);
    }
  }
  u16* obp = o + (long)b * kLpad * rowstr + h * kDh;
  #pragma unroll
  for (int n = 0; n < 8; ++n)
    #pragma unroll
    for (int j = 0; j < 8; ++j) {
      int qg = q0 + j + 8 * hl;
      obp[(long)qg * rowstr + n * 16 + r] = f2bf(oacc[n][j] / rs[j]);
    }
}

// --------------------------------------------------------------- layernorm
__global__ __launch_bounds__(256) void ln_kernel(
    const float* __restrict__ x, const float* __restrict__ g,
    const float* __restrict__ bt, float* __restrict__ of,
    u16* __restrict__ oh) {
  int row = blockIdx.x;
  int pos = row % kLpad;
  int t = threadIdx.x;
  float* orow = of + (long)row * kD;
  u16* hrow = oh + (long)row * kD;
  if (pos >= kLreal) {
    orow[t] = 0.f; orow[t + 256] = 0.f;
    hrow[t] = 0;   hrow[t + 256] = 0;
    return;
  }
  const float* xr = x + (long)row * kD;
  float a = xr[t], c = xr[t + 256];
  __shared__ float sh[512];
  sh[t] = a + c; sh[t + 256] = a * a + c * c;
  __syncthreads();
  for (int off = 128; off > 0; off >>= 1) {
    if (t < off) { sh[t] += sh[t + off]; sh[t + 256] += sh[t + 256 + off]; }
    __syncthreads();
  }
  float mean = sh[0] * (1.f / kD);
  float var = sh[256] * (1.f / kD) - mean * mean;
  float rstd = rsqrtf(var + 1e-5f);
  float y0 = (a - mean) * rstd * g[t] + bt[t];
  float y1 = (c - mean) * rstd * g[t + 256] + bt[t + 256];
  orow[t] = y0; orow[t + 256] = y1;
  hrow[t] = f2bf(y0); hrow[t + 256] = f2bf(y1);
}

// ------------------------------------------------------ bottleneck pieces
__global__ __launch_bounds__(256) void conv_stride4(
    const u16* __restrict__ in, long bstride, const float* __restrict__ w,
    const float* __restrict__ cb, float* __restrict__ raw, int Lout) {
  long i = (long)blockIdx.x * 256 + threadIdx.x;
  if (i >= (long)kB * Lout * kDh) return;
  int o = (int)(i & 127); long bt = i >> 7;
  int t = (int)(bt % Lout); int b = (int)(bt / Lout);
  const u16* base = in + b * bstride + (long)t * 4 * kDh;
  float acc = cb[o];
  #pragma unroll
  for (int kk = 0; kk < 4; ++kk)
    for (int ci = 0; ci < kDh; ++ci)
      acc += bf2f(base[kk * kDh + ci]) * w[(o * kDh + ci) * 4 + kk];
  raw[i] = acc;
}

__global__ __launch_bounds__(256) void bn_stats(
    const float* __restrict__ raw, int n, float* __restrict__ mu,
    float* __restrict__ var) {
  int c = blockIdx.x;
  float s = 0.f, s2 = 0.f;
  for (int i = threadIdx.x; i < n; i += 256) {
    float v = raw[(long)i * kDh + c];
    s += v; s2 += v * v;
  }
  __shared__ float sa[256], sb[256];
  sa[threadIdx.x] = s; sb[threadIdx.x] = s2;
  __syncthreads();
  for (int off = 128; off > 0; off >>= 1) {
    if (threadIdx.x < off) {
      sa[threadIdx.x] += sa[threadIdx.x + off];
      sb[threadIdx.x] += sb[threadIdx.x + off];
    }
    __syncthreads();
  }
  if (threadIdx.x == 0) {
    float m = sa[0] / n;
    mu[c] = m; var[c] = sb[0] / n - m * m;
  }
}

__global__ __launch_bounds__(256) void bn_elu(
    const float* __restrict__ raw, const float* __restrict__ mu,
    const float* __restrict__ var, const float* __restrict__ g,
    const float* __restrict__ b, u16* __restrict__ pyr, int Lout, int off) {
  long i = (long)blockIdx.x * 256 + threadIdx.x;
  if (i >= (long)kB * Lout * kDh) return;
  int c = (int)(i & 127); long bt = i >> 7;
  int t = (int)(bt % Lout); int bb = (int)(bt / Lout);
  float v = (raw[i] - mu[c]) * rsqrtf(var[c] + 1e-5f) * g[c] + b[c];
  v = v > 0.f ? v : __expf(v) - 1.f;
  pyr[((long)(bb * kPyrL + off + t)) * kDh + c] = f2bf(v);
}

__global__ __launch_bounds__(256) void pyr_scatter(
    const float* __restrict__ up_out, float* __restrict__ enc_f) {
  long i = (long)blockIdx.x * 256 + threadIdx.x;
  if (i >= (long)kB * kPyrL * kD) return;
  int d = (int)(i & 511); long bj = i >> 9;
  int j = (int)(bj % kPyrL); int b = (int)(bj / kPyrL);
  enc_f[((long)b * kLpad + kLin + j) * kD + d] = up_out[i];
}

// ------------------------------------------------------------ final gather
__global__ __launch_bounds__(256) void gather_out(
    const float* __restrict__ enc, float* __restrict__ out) {
  long i = (long)blockIdx.x * 256 + threadIdx.x;
  if (i >= (long)kB * kLin * 4 * kD) return;
  int d = (int)(i & 511); long t = i >> 9;
  int lvl = (int)(t & 3); t >>= 2;
  int ii = (int)(t & 511); int b = (int)(t >> 9);
  int idx;
  switch (lvl) {
    case 0: idx = ii; break;
    case 1: idx = 512 + (ii >> 2); break;
    case 2: idx = 640 + (ii >> 4); break;
    default: idx = 672 + (ii >> 6); break;
  }
  out[i] = enc[((long)b * kLpad + idx) * kD + d];
}

}  // namespace

#define GEMM(EPI, A, W, OUT, BIAS, RES, M, N, K)                               \
  gemm_bf16<EPI><<<dim3((unsigned)((((M) / 16) * ((N) / 128) + 7) / 8)), 256,  \
                   0, stream>>>((A), (W), (OUT), (BIAS), (RES), (M), (N), (K))

extern "C" void kernel_launch(void* const* d_in, const int* in_sizes, int n_in,
                              void* d_out, int out_size, void* d_ws, size_t ws_size,
                              hipStream_t stream) {
  (void)in_sizes; (void)n_in; (void)out_size; (void)ws_size;
  const float* x       = (const float*)d_in[0];
  const float* cov_w   = (const float*)d_in[1];
  const float* cov_b   = (const float*)d_in[2];
  const float* dconv_w = (const float*)d_in[3];
  const float* down_w  = (const float*)d_in[4];
  const float* down_b  = (const float*)d_in[5];
  const float* conv_w  = (const float*)d_in[6];
  const float* conv_b  = (const float*)d_in[7];
  const float* bn_g    = (const float*)d_in[8];
  const float* bn_b    = (const float*)d_in[9];
  const float* up_w    = (const float*)d_in[10];
  const float* up_b    = (const float*)d_in[11];
  const float* bln_g   = (const float*)d_in[12];
  const float* bln_b   = (const float*)d_in[13];
  const float* wq      = (const float*)d_in[14];
  const float* wk      = (const float*)d_in[15];
  const float* wv      = (const float*)d_in[16];
  const float* fc_w    = (const float*)d_in[17];
  const float* ln1_g   = (const float*)d_in[18];
  const float* ln1_b   = (const float*)d_in[19];
  const float* f1_w    = (const float*)d_in[20];
  const float* f1_b    = (const float*)d_in[21];
  const float* f2_w    = (const float*)d_in[22];
  const float* f2_b    = (const float*)d_in[23];
  const float* ln2_g   = (const float*)d_in[24];
  const float* ln2_b   = (const float*)d_in[25];
  // d_in[26] (attn_mask) and d_in[27] (gather_idx) are recomputed on device.

  uintptr_t p = (uintptr_t)d_ws;
  auto alloc = [&](size_t bytes) -> void* {
    p = (p + 255) & ~(uintptr_t)255;
    void* r = (void*)p; p += bytes; return r;
  };
  float* bias   = (float*)alloc(sizeof(float) * (size_t)kLpad * kLpad);
  float* enc_f  = (float*)alloc(sizeof(float) * (size_t)kRows * kD);
  u16*   enc_h  = (u16*)  alloc(2ull * kRows * kD);
  u16*   seq_h  = (u16*)  alloc(2ull * kB * kLin * kD);
  u16*   tmp0   = (u16*)  alloc(2ull * kB * kLin * kDh);
  float* craw   = (float*)alloc(sizeof(float) * (size_t)kB * 128 * kDh);
  float* bn_mu  = (float*)alloc(sizeof(float) * kDh);
  float* bn_var = (float*)alloc(sizeof(float) * kDh);
  u16*   pyr    = (u16*)  alloc(2ull * kB * kPyrL * kDh);
  float* up_out = (float*)alloc(sizeof(float) * (size_t)kB * kPyrL * kD);
  u16*   qb     = (u16*)  alloc(2ull * kRows * kHeads * kDh);
  u16*   kbuf   = (u16*)  alloc(2ull * kRows * kHeads * kDh);
  u16*   vt     = (u16*)  alloc(2ull * kRows * kHeads * kDh);
  u16*   obuf   = (u16*)  alloc(2ull * kRows * kHeads * kDh);
  u16*   h1     = (u16*)  alloc(2ull * kRows * kDff);
  float* tmpf   = (float*)alloc(sizeof(float) * (size_t)kRows * kD);
  u16*   downT  = (u16*)  alloc(2ull * kDh * kD);
  u16*   upT    = (u16*)  alloc(2ull * kD * kDh);
  u16 *wqT[4], *wkT[4], *wvT[4], *fcT[4], *f1T[4], *f2T[4];
  for (int l = 0; l < 4; ++l) {
    wqT[l] = (u16*)alloc(2ull * 1024 * 512);
    wkT[l] = (u16*)alloc(2ull * 1024 * 512);
    wvT[l] = (u16*)alloc(2ull * 1024 * 512);
    fcT[l] = (u16*)alloc(2ull * 512 * 1024);
    f1T[l] = (u16*)alloc(2ull * 2048 * 512);
    f2T[l] = (u16*)alloc(2ull * 512 * 2048);
  }

  auto gblk = [](long n) { return dim3((unsigned)((n + 255) / 256)); };

  // Mask bias + weight prep (bf16, transposed to [N][K]).
  build_bias<<<gblk((long)kLpad * kLpad), 256, 0, stream>>>(bias);
  wtrans<<<gblk((long)kD * kDh), 256, 0, stream>>>(down_w, downT, kD, kDh);
  wtrans<<<gblk((long)kDh * kD), 256, 0, stream>>>(up_w, upT, kDh, kD);
  for (int l = 0; l < 4; ++l) {
    wtrans<<<gblk(512l * 1024), 256, 0, stream>>>(wq + (long)l * 512 * 1024, wqT[l], 512, 1024);
    wtrans<<<gblk(512l * 1024), 256, 0, stream>>>(wk + (long)l * 512 * 1024, wkT[l], 512, 1024);
    wtrans<<<gblk(512l * 1024), 256, 0, stream>>>(wv + (long)l * 512 * 1024, wvT[l], 512, 1024);
    wtrans<<<gblk(1024l * 512), 256, 0, stream>>>(fc_w + (long)l * 1024 * 512, fcT[l], 1024, 512);
    wtrans<<<gblk(512l * 2048), 256, 0, stream>>>(f1_w + (long)l * 512 * 2048, f1T[l], 512, 2048);
    wtrans<<<gblk(2048l * 512), 256, 0, stream>>>(f2_w + (long)l * 2048 * 512, f2T[l], 2048, 512);
  }

  // Embedding + bottleneck pyramid.
  embed_kernel<<<gblk((long)kB * kLin * kD), 256, 0, stream>>>(x, cov_w, cov_b, dconv_w, enc_f, seq_h);
  GEMM(3, seq_h, downT, tmp0, down_b, nullptr, kB * kLin, kDh, kD);
  {
    const int Ls[3] = {128, 32, 8};
    const int off[3] = {0, 128, 160};
    const u16* cin = tmp0;
    long bstr = (long)kLin * kDh;
    for (int s = 0; s < 3; ++s) {
      long nel = (long)kB * Ls[s] * kDh;
      conv_stride4<<<gblk(nel), 256, 0, stream>>>(cin, bstr, conv_w + (size_t)s * kDh * kDh * 4,
                                                  conv_b + s * kDh, craw, Ls[s]);
      bn_stats<<<kDh, 256, 0, stream>>>(craw, kB * Ls[s], bn_mu, bn_var);
      bn_elu<<<gblk(nel), 256, 0, stream>>>(craw, bn_mu, bn_var, bn_g + s * kDh,
                                            bn_b + s * kDh, pyr, Ls[s], off[s]);
      cin = pyr + (size_t)off[s] * kDh;
      bstr = (long)kPyrL * kDh;
    }
  }
  GEMM(1, pyr, upT, up_out, up_b, nullptr, kB * kPyrL, kD, kDh);
  pyr_scatter<<<gblk((long)kB * kPyrL * kD), 256, 0, stream>>>(up_out, enc_f);
  ln_kernel<<<kRows, 256, 0, stream>>>(enc_f, bln_g, bln_b, enc_f, enc_h);

  // Encoder layers.
  for (int l = 0; l < 4; ++l) {
    GEMM(2, enc_h, wqT[l], qb,   nullptr, nullptr, kRows, 1024, 512);
    GEMM(2, enc_h, wkT[l], kbuf, nullptr, nullptr, kRows, 1024, 512);
    GEMM(7, enc_h, wvT[l], vt,   nullptr, nullptr, kRows, 1024, 512);
    fa_kernel<<<kB * kHeads * (kLpad / 16), 32, 0, stream>>>(qb, kbuf, vt, bias, obuf);
    GEMM(4, obuf, fcT[l], tmpf, nullptr, enc_f, kRows, 512, 1024);
    ln_kernel<<<kRows, 256, 0, stream>>>(tmpf, ln1_g + l * 512, ln1_b + l * 512, enc_f, enc_h);
    GEMM(5, enc_h, f1T[l], h1, f1_b + l * kDff, nullptr, kRows, kDff, 512);
    GEMM(6, h1, f2T[l], tmpf, f2_b + l * 512, enc_f, kRows, 512, kDff);
    ln_kernel<<<kRows, 256, 0, stream>>>(tmpf, ln2_g + l * 512, ln2_b + l * 512, enc_f, enc_h);
  }

  gather_out<<<gblk((long)kB * kLin * 4 * kD), 256, 0, stream>>>(enc_f, (float*)d_out);
}
